// VOneBlock_86191403696594
// MI455X (gfx1250) — matrix operand order repaired and checked by reading, not compiled, not run on hardware
//
#include <hip/hip_runtime.h>
#include <hip/hip_bf16.h>
#include <cstdint>
#include <cstddef>

typedef __attribute__((ext_vector_type(16))) _Float16     v16h;
typedef __attribute__((ext_vector_type(8)))  float        v8f;
typedef __attribute__((ext_vector_type(4)))  unsigned int v4u;

// ---------------- problem constants ----------------
constexpr int B_   = 32,  CIN = 3,  H = 224, W = 224;
constexpr int SC   = 256, OC  = 512;
constexpr int KS   = 25,  STRIDE = 4, PAD = 12;
constexpr int HO   = 56,  WO = 56;
constexpr int HW   = HO * WO;            // 3136
constexpr int NTOT = B_ * HW;            // 100352
constexpr int KREAL = CIN * KS * KS;     // 1875
constexpr int KP    = 1888;              // 59 * 32 (zero padded)
constexpr int MROWS = OC + SC;           // 768 (512 w0 rows + 256 w1 rows)
constexpr int BM = 128, BN = 128, BK = 32;   // BN = two (b,oh) rows of 64 cols (56 live)
constexpr int AST = 40;                  // LDS row stride (halves): bank-conflict-free, 16B aligned
constexpr int BST = 40;
constexpr int WQ  = 56;                  // phase-plane width (224/4)

constexpr size_t WH_BYTES = (size_t)MROWS * KP * 2;            // 2,899,968
constexpr int    XN       = B_ * CIN * H * W;                  // 4,816,896
constexpr size_t XH_BYTES = (size_t)XN * 2;                    // 9,633,792

union Frag { v4u u[2]; v16h h; };

// ---------------- weight packing: [768][1888] f16, row-major ----------------
__global__ __launch_bounds__(256) void pack_w(const float* __restrict__ w0,
                                              const float* __restrict__ w1,
                                              _Float16* __restrict__ wh) {
    int idx = blockIdx.x * 256 + threadIdx.x;
    if (idx >= MROWS * KP) return;
    int r = idx / KP;
    int k = idx - r * KP;
    float v = 0.0f;
    if (k < KREAL) {
        v = (r < OC) ? w0[(size_t)r * KREAL + k]
                     : w1[(size_t)(r - SC) * KREAL + k];
    }
    wh[idx] = (_Float16)v;
}

// -------- input: f32 -> f16 with stride-4 phase split --------
// xp[p][b][cin][ih][wq] = x[b][cin][ih][4*wq + p]
__global__ __launch_bounds__(256) void pack_x_phase(const float* __restrict__ x,
                                                    _Float16* __restrict__ xp) {
    int i = blockIdx.x * 256 + threadIdx.x;
    if (i >= XN) return;
    int wq = i % WQ;  int t = i / WQ;
    int ih = t % H;   t /= H;
    int cin = t % CIN; t /= CIN;
    int b  = t % B_;  int p = t / B_;
    float v = x[((size_t)(b * CIN + cin) * H + ih) * W + wq * 4 + p];
    xp[i] = (_Float16)v;
}

// ---------------- implicit-GEMM conv via v_wmma_f32_16x16x32_f16 ----------------
// One block = (b, oh0..oh0+1) output rows x 128 channel rows.
// S[row][n] row-major, n = b*HW + oh*WO + ow.
__global__ __launch_bounds__(256) void gemm_wmma(const _Float16* __restrict__ Wh,
                                                 const _Float16* __restrict__ Xp,
                                                 float* __restrict__ S) {
    __shared__ __align__(16) _Float16 As[2][BM * AST];   // [m][k], padded rows
    __shared__ __align__(16) _Float16 Bs[2][BN * BST];   // [n][k], padded rows (n: 2x64)

    const int tid = threadIdx.x;
    const int m0  = blockIdx.y * BM;
    const int bx  = blockIdx.x;              // 0 .. B_*HO/2 - 1
    const int b   = bx / (HO / 2);
    const int oh0 = (bx - b * (HO / 2)) * 2;

    // ---- A staging role: 16 halves (32B) per thread, contiguous
    const int ar = tid >> 1;                 // 0..127
    const int ac = (tid & 1) * 16;           // 0 / 16
    const _Float16* wrow = Wh + (size_t)(m0 + ar) * KP + ac;

    // ---- B staging role: one k-row, 16 consecutive n (stays within one oh-row)
    const int bk  = tid >> 3;                // 0..31  (k within chunk)
    const int bn0 = (tid & 7) * 16;          // 0..112 (n base)
    const int soh = oh0 + (bn0 >> 6);        // this thread's oh row
    const int sow0 = bn0 & 63;               // ow base within row
    const int ihbase = soh * STRIDE - PAD;

    // ---- wave mapping: 8 waves -> 2(M) x 4(N), wave tile 64x32
    const int wid   = tid >> 5;
    const int lane  = tid & 31;
    const int mw    = (wid >> 2) * 64;       // 0 / 64
    const int nwave = (wid & 3) * 32;        // 0,32,64,96
    const int lr    = lane & 15;
    const int lh    = lane >> 4;

    v8f acc[4][2] = {};

    v4u a0, a1;
    _Float16 bh[16];

    auto fetch = [&](int kc) {
        { // A: vector global load
            const v4u* g = reinterpret_cast<const v4u*>(wrow + kc);
            a0 = g[0]; a1 = g[1];
        }
        // B: decode k ONCE, then contiguous phase-plane span with clipping
        int  k    = kc + bk;
        bool kval = k < KREAL;
        int  kk   = kval ? k : 0;
        int  cin  = kk / 625;
        int  rem  = kk - cin * 625;
        int  kh   = rem / 25;
        int  kw   = rem - kh * 25;
        int  ih   = ihbase + kh;
        bool ihok = kval && ((unsigned)ih < (unsigned)H);
        int  ihc  = ihok ? ih : 0;
        int  t4   = kw + 4;                  // kw - 12 = t4 - 16
        int  p    = t4 & 3;
        int  q0   = (t4 >> 2) - 4 + sow0;    // wq of first element
        const _Float16* rowp =
            Xp + ((((size_t)p * B_ + b) * CIN + cin) * H + ihc) * WQ;
#pragma unroll
        for (int j = 0; j < 16; ++j) {
            int  wq = q0 + j;
            bool ok = ihok && ((unsigned)wq < (unsigned)WQ);
            bh[j] = ok ? rowp[wq] : (_Float16)0.0f;
        }
    };

    auto stage = [&](int buf) {
        v4u* d = reinterpret_cast<v4u*>(&As[buf][ar * AST + ac]);
        d[0] = a0; d[1] = a1;
#pragma unroll
        for (int j = 0; j < 16; ++j)
            Bs[buf][(bn0 + j) * BST + bk] = bh[j];    // transposed scatter, conflict-free
    };

    fetch(0);
    constexpr int NC = KP / BK;              // 59
    for (int c = 0; c < NC; ++c) {
        const int buf = c & 1;
        stage(buf);
        __syncthreads();
        if (c + 1 < NC) fetch((c + 1) * BK); // global loads overlap WMMA burst below

        // load ALL fragments first -> scheduler can relax ds waits
        Frag bf[2];
#pragma unroll
        for (int nt = 0; nt < 2; ++nt) {     // B 32x16: halves {g..g+7, g+16..g+23}, g=lh*8
            const v4u* bp = reinterpret_cast<const v4u*>(
                &Bs[buf][(nwave + nt * 16 + lr) * BST + lh * 8]);
            bf[nt].u[0] = bp[0]; bf[nt].u[1] = bp[2];
        }
        Frag af[4];
#pragma unroll
        for (int mt = 0; mt < 4; ++mt) {     // A 16x32: halves {g..g+7, g+16..g+23}, g=lh*8
            const _Float16* ap = &As[buf][(mw + mt * 16 + lr) * AST + lh * 8];
            af[mt].u[0] = *reinterpret_cast<const v4u*>(ap);
            af[mt].u[1] = *reinterpret_cast<const v4u*>(ap + 16);
        }
#pragma unroll
        for (int mt = 0; mt < 4; ++mt)
#pragma unroll
            for (int nt = 0; nt < 2; ++nt)
                acc[mt][nt] = __builtin_amdgcn_wmma_f32_16x16x32_f16(
                    false, af[mt].h, false, bf[nt].h, (short)0, acc[mt][nt], false, false);
    }

    // ---- epilogue: C layout VGPR r -> M = r + 8*lh, N = lr; skip dead columns
#pragma unroll
    for (int nt = 0; nt < 2; ++nt) {
        const int col = nwave + nt * 16 + lr;     // 0..127
        const int ow  = col & 63;
        if (ow < WO) {
            const int oh = oh0 + (col >> 6);
            const size_t obase = (size_t)b * HW + (size_t)oh * WO + ow;
#pragma unroll
            for (int mt = 0; mt < 4; ++mt) {
                int mbase = m0 + mw + mt * 16 + lh * 8;
#pragma unroll
                for (int r = 0; r < 8; ++r)
                    S[(size_t)(mbase + r) * NTOT + obase] = acc[mt][nt][r];
            }
        }
    }
}

// ---------------- pointwise + instance norm + relu, one block per (b,c) ----------------
__global__ __launch_bounds__(256) void fuse_norm(const float* __restrict__ S,
                                                 float* __restrict__ out) {
    const int bc = blockIdx.x;        // 0..16383
    const int b  = bc >> 9;           // /512
    const int c  = bc & 511;
    const int t  = threadIdx.x;

    const size_t nbase = (size_t)b * HW;
    const float* S0 = S + (size_t)c * NTOT + nbase;
    const float* S1 = S + (size_t)(c + SC) * NTOT + nbase;  // used only when c>=SC

    float vals[13];
    int   cnt = 0;
    float sum = 0.0f, sumsq = 0.0f;
    for (int i = t; i < HW; i += 256) {
        float v;
        if (c < SC) {
            v = 25.0f * fmaxf(S0[i], 0.0f);
        } else {
            float s0 = S0[i], s1 = S1[i];
            float cv  = 25.0f * sqrtf(s0 * s0 + s1 * s1) * 0.70710678118654752f;
            float net = 3.0f * cv;                    // xc + DELTA*xc, DELTA=2
            v = 10.0f * net / (0.2f + net);           // BETA*net/(ALPHA+net)
        }
        vals[cnt++] = v;
        sum += v;
        sumsq += v * v;
    }

    __shared__ float rs[256];
    __shared__ float rq[256];
    rs[t] = sum; rq[t] = sumsq;
    __syncthreads();
    for (int s = 128; s > 0; s >>= 1) {
        if (t < s) { rs[t] += rs[t + s]; rq[t] += rq[t + s]; }
        __syncthreads();
    }
    const float inv  = 1.0f / (float)HW;
    const float mean = rs[0] * inv;
    const float var  = rq[0] * inv - mean * mean;
    const float rstd = rsqrtf(var + 1e-5f);

    float* O = out + (size_t)bc * HW;
    cnt = 0;
    for (int i = t; i < HW; i += 256) {
        float v = (vals[cnt++] - mean) * rstd;
        O[i] = fmaxf(v, 0.0f);
    }
}

// ---------------- launcher ----------------
extern "C" void kernel_launch(void* const* d_in, const int* in_sizes, int n_in,
                              void* d_out, int out_size, void* d_ws, size_t ws_size,
                              hipStream_t stream) {
    (void)in_sizes; (void)n_in; (void)out_size; (void)ws_size;
    const float* x  = (const float*)d_in[0];
    const float* w0 = (const float*)d_in[1];
    const float* w1 = (const float*)d_in[2];
    float* out = (float*)d_out;

    char* ws = (char*)d_ws;
    _Float16* wh = (_Float16*)(ws);
    _Float16* xp = (_Float16*)(ws + WH_BYTES);
    float*    S  = (float*)   (ws + WH_BYTES + XH_BYTES);   // 768 x 100352 f32

    pack_w<<<(MROWS * KP + 255) / 256, 256, 0, stream>>>(w0, w1, wh);
    pack_x_phase<<<(XN + 255) / 256, 256, 0, stream>>>(x, xp);

    dim3 grid(B_ * (HO / 2), MROWS / BM);   // (896, 6)
    gemm_wmma<<<grid, 256, 0, stream>>>(wh, xp, S);

    fuse_norm<<<B_ * OC, 256, 0, stream>>>(S, out);
}
